// SparseConv2d_15058155340244
// MI455X (gfx1250) — compile-verified, matrix operand order checked by reading, not run on hardware
//
#include <hip/hip_runtime.h>

// CDNA5 (gfx1250) wave32 WMMA types
typedef __attribute__((ext_vector_type(16))) __bf16 v16bf;
typedef __attribute__((ext_vector_type(8)))  float  v8f;

#define IN_C   32
#define OUT_C  32
#define HW     256
#define TILE_W 64
#define COLS   (TILE_W + 2)          // 66 input columns incl. halo
#define WAVES  4
#define BLOCK  (WAVES * 32)          // 128 threads, 4 wave32
#define NTAP   9
#define WELEMS (NTAP * OUT_C * IN_C) // 9216 bf16 per (hi|lo) plane
#define WBYTES (2u * WELEMS * (unsigned)sizeof(__bf16)) // 36864 B in d_ws

// Load one 16x32 bf16 WMMA fragment for this lane: two contiguous 16B chunks,
// K = {ic0..ic0+7} then {ic0+16..ic0+23} (ISA 16-bit A/B VGPR layout).
__device__ __forceinline__ v16bf load_frag(const __bf16* p) {
    union { v16bf v; uint4 q[2]; } f;
    f.q[0] = *(const uint4*)(p);
    f.q[1] = *(const uint4*)(p + 16);
    return f.v;
}

__device__ __forceinline__ v8f wmma_bf16(v16bf a, v16bf b, v8f c) {
    return __builtin_amdgcn_wmma_f32_16x16x32_bf16(
        /*neg_a=*/false, a, /*neg_b=*/false, b,
        /*c_mod=*/(short)0, c, /*reuse_a=*/false, /*reuse_b=*/false);
}

// One-shot: convert weights [oc][ic][kh][kw] f32 -> pre-swizzled bf16 hi|lo
// planes [tap][oc][ic] in global scratch (stays L2-resident: 36 KB).
__global__ void prep_weights(const float* __restrict__ w, __bf16* __restrict__ whl) {
    const int idx = blockIdx.x * 256 + threadIdx.x;
    if (idx >= WELEMS) return;
    const int oc  = idx / (IN_C * NTAP);
    const int rem = idx - oc * (IN_C * NTAP);
    const int ic  = rem / NTAP;
    const int khw = rem - ic * NTAP;             // idx == oc*288 + ic*9 + khw
    const float v = w[idx];
    const __bf16 h = (__bf16)v;                  // RNE
    const __bf16 l = (__bf16)(v - (float)h);     // residual
    whl[(khw * OUT_C + oc) * IN_C + ic]          = h;
    whl[WELEMS + (khw * OUT_C + oc) * IN_C + ic] = l;
}

// Pack 4 f32 -> 4 bf16 hi parts + 4 bf16 lo parts (as uint2 each)
union pack4 { __bf16 e[4]; uint2 u; };

__global__ __launch_bounds__(BLOCK)
void conv3x3_bf16x3_wmma(const float* __restrict__ x,
                         const float* __restrict__ w,
                         const float* __restrict__ bias,
                         const __bf16* __restrict__ whl,
                         float* __restrict__ y,
                         int prew) {
    // Input tile, bf16 hi/lo planes, layout [row(3)][col(66)][ic(32)] (ic innermost)
    __shared__ alignas(16) __bf16 in_hi[3 * COLS * IN_C];
    __shared__ alignas(16) __bf16 in_lo[3 * COLS * IN_C];
    // Weights, bf16 hi/lo, layout [tap(9)][oc(32)][ic(32)]
    __shared__ alignas(16) __bf16 w_hi[WELEMS];
    __shared__ alignas(16) __bf16 w_lo[WELEMS];

    const int tid = threadIdx.x;
    const int w0  = blockIdx.x * TILE_W;  // first output column of strip
    const int oh  = blockIdx.y;           // output row
    const int nb  = blockIdx.z;           // batch

    // ---- Stage weights into LDS (uniform branch: one path per launch) ----
    if (prew) {
        // straight vector copy of pre-swizzled bf16 planes (9 iters, no divides)
        const uint4* src = (const uint4*)whl;
        uint4* dh = (uint4*)w_hi;
        uint4* dl = (uint4*)w_lo;
        const int NV = WELEMS / 8;               // 1152 uint4 per plane
        for (int i = tid; i < NV; i += BLOCK) {
            dh[i] = src[i];
            dl[i] = src[NV + i];
        }
    } else {
        // fallback: convert in-kernel (only if scratch too small)
        for (int idx = tid; idx < WELEMS; idx += BLOCK) {
            const int oc  = idx / (IN_C * NTAP);
            const int rem = idx - oc * (IN_C * NTAP);
            const int ic  = rem / NTAP;
            const int khw = rem - ic * NTAP;
            const float v = w[idx];
            const __bf16 h = (__bf16)v;
            const __bf16 l = (__bf16)(v - (float)h);
            w_hi[(khw * OUT_C + oc) * IN_C + ic] = h;
            w_lo[(khw * OUT_C + oc) * IN_C + ic] = l;
        }
    }

    // ---- Stage input halo (rows oh-1..oh+1, cols w0-1..w0+64) as bf16 hi/lo ----
    // Each thread owns up to two (row,col) positions, decomposed ONCE; inner
    // loop covers 4 channels/iter, packing 4 bf16 into one ds_store_b64 per plane.
    const float* xn = x + (size_t)nb * IN_C * HW * HW;
    {
        const int pos0 = tid;                    // 0..127  (< 198 always)
        const int r0   = pos0 / COLS;
        const int c0   = pos0 - r0 * COLS;
        const int pos1 = tid + BLOCK;            // 128..255
        const int r1   = pos1 / COLS;
        const int c1   = pos1 - r1 * COLS;
        const bool has1 = (pos1 < 3 * COLS);

        const int gh0 = oh + r0 - 1, gw0 = w0 + c0 - 1;
        const int gh1 = oh + r1 - 1, gw1 = w0 + c1 - 1;
        const bool ok0 = ((unsigned)gh0 < HW) && ((unsigned)gw0 < HW);
        const bool ok1 = has1 && ((unsigned)gh1 < HW) && ((unsigned)gw1 < HW);
        const long  go0 = (long)gh0 * HW + gw0;  // may be negative when !ok0 (never read)
        const long  go1 = (long)gh1 * HW + gw1;
        const int   lb0 = (r0 * COLS + c0) * IN_C;
        const int   lb1 = (r1 * COLS + c1) * IN_C;

        for (int ic = 0; ic < IN_C; ic += 4) {
            pack4 h0, l0, h1, l1;
            #pragma unroll
            for (int j = 0; j < 4; ++j) {
                const long chan = (long)(ic + j) * (HW * HW);
                const float v0 = ok0 ? xn[chan + go0] : 0.0f;
                h0.e[j] = (__bf16)v0;
                l0.e[j] = (__bf16)(v0 - (float)h0.e[j]);
                const float v1 = ok1 ? xn[chan + go1] : 0.0f;
                h1.e[j] = (__bf16)v1;
                l1.e[j] = (__bf16)(v1 - (float)h1.e[j]);
            }
            *(uint2*)&in_hi[lb0 + ic] = h0.u;    // 8B-aligned (lb0 is 32-elem mult)
            *(uint2*)&in_lo[lb0 + ic] = l0.u;
            if (has1) {
                *(uint2*)&in_hi[lb1 + ic] = h1.u;
                *(uint2*)&in_lo[lb1 + ic] = l1.u;
            }
        }
    }

    __syncthreads();

    // ---- Per-wave implicit GEMM: 32(oc) x 16(pixels), K = ic per tap ----
    const int lane    = tid & 31;
    const int n       = lane & 15;       // pixel / A-row index
    const int half    = lane >> 4;       // lane group (selects K block)
    const int ic0     = half * 8;        // K base: {0..7,16..23} or {8..15,24..31}
    const int pixbase = (tid >> 5) * 16; // this wave's pixel strip

    v8f acc0 = {};  // oc 0..15
    v8f acc1 = {};  // oc 16..31

    for (int khw = 0; khw < NTAP; ++khw) {
        const int kh = khw / 3;
        const int kw = khw - 3 * kh;

        // B fragments (activations): K=ic contiguous in LDS
        const int bcol = pixbase + n + kw;             // 0..65
        const v16bf Bh = load_frag(&in_hi[(kh * COLS + bcol) * IN_C + ic0]);
        const v16bf Bl = load_frag(&in_lo[(kh * COLS + bcol) * IN_C + ic0]);

        // A fragments (weights), two oc tiles
        const v16bf Ah0 = load_frag(&w_hi[(khw * OUT_C + n)      * IN_C + ic0]);
        const v16bf Al0 = load_frag(&w_lo[(khw * OUT_C + n)      * IN_C + ic0]);
        const v16bf Ah1 = load_frag(&w_hi[(khw * OUT_C + 16 + n) * IN_C + ic0]);
        const v16bf Al1 = load_frag(&w_lo[(khw * OUT_C + 16 + n) * IN_C + ic0]);

        // bf16x3: D += Ah*Bh + Ah*Bl + Al*Bh  (f32 accumulate)
        acc0 = wmma_bf16(Ah0, Bh, acc0);
        acc0 = wmma_bf16(Ah0, Bl, acc0);
        acc0 = wmma_bf16(Al0, Bh, acc0);
        acc1 = wmma_bf16(Ah1, Bh, acc1);
        acc1 = wmma_bf16(Ah1, Bl, acc1);
        acc1 = wmma_bf16(Al1, Bh, acc1);
    }

    // ---- Epilogue: D layout -> lane(half,n), VGPR r holds M = half*8 + r ----
    // Vector-load the 8 bias values each accumulator tile needs (2x float4).
    union { float4 v[2]; float f[8]; } b0, b1;
    b0.v[0] = ((const float4*)bias)[half * 2 + 0];      // bias[half*8 .. +3]
    b0.v[1] = ((const float4*)bias)[half * 2 + 1];      // bias[half*8+4 .. +7]
    b1.v[0] = ((const float4*)bias)[4 + half * 2 + 0];  // bias[16+half*8 ..]
    b1.v[1] = ((const float4*)bias)[4 + half * 2 + 1];

    const int gw = w0 + pixbase + n;
    float* yp0 = y + (((size_t)nb * OUT_C + half * 8)      * HW + oh) * HW + gw;
    float* yp1 = y + (((size_t)nb * OUT_C + 16 + half * 8) * HW + oh) * HW + gw;
    #pragma unroll
    for (int r = 0; r < 8; ++r) {
        yp0[(size_t)r * HW * HW] = acc0[r] + b0.f[r];   // imm-offset stores
        yp1[(size_t)r * HW * HW] = acc1[r] + b1.f[r];
    }
}

extern "C" void kernel_launch(void* const* d_in, const int* in_sizes, int n_in,
                              void* d_out, int out_size, void* d_ws, size_t ws_size,
                              hipStream_t stream) {
    const float* x = (const float*)d_in[0];
    const float* w = (const float*)d_in[1];
    const float* b = (const float*)d_in[2];
    float*       y = (float*)d_out;

    const int N = in_sizes[0] / (IN_C * HW * HW);   // 16
    dim3 grid(HW / TILE_W, HW, N);                  // (4, 256, 16)

    const int prew = (ws_size >= (size_t)WBYTES) ? 1 : 0;
    __bf16* whl = (__bf16*)d_ws;
    if (prew)
        prep_weights<<<(WELEMS + 255) / 256, 256, 0, stream>>>(w, whl);
    conv3x3_bf16x3_wmma<<<grid, BLOCK, 0, stream>>>(x, w, b, whl, y, prew);
    (void)n_in; (void)out_size;
}
